// InterleavedHiddenMarkovChain_19756849562182
// MI455X (gfx1250) — compile-verified, matrix operand order checked by reading, not run on hardware
//
#include <hip/hip_runtime.h>
#include <hip/hip_bf16.h>
#include <math.h>

typedef __attribute__((ext_vector_type(16))) _Float16 v16h;
typedef __attribute__((ext_vector_type(8)))  _Float16 v8h;
typedef __attribute__((ext_vector_type(8)))  float    v8f;

#define KCH   3
#define NS    12
#define ALPH  32
#define TLEN  64
#define NTUP  1728          // 12^3
#define NCOMP 5184          // 12^3 * 3
#define BLOCK 864           // 27 waves of 32: one wave per (chain, col-tile)
#define BQ_STRIDE 4608      // 144 cols * 32 padded K (halves) per chain
#define GP_STRIDE 2304      // 16 rows * 144 cols (floats) per chain

__device__ __forceinline__ float waveMax(float v) {
    #pragma unroll
    for (int off = 16; off > 0; off >>= 1)
        v = fmaxf(v, __shfl_xor(v, off, 32));
    return v;
}
__device__ __forceinline__ float waveSum(float v) {
    #pragma unroll
    for (int off = 16; off > 0; off >>= 1)
        v += __shfl_xor(v, off, 32);
    return v;
}

__global__ __launch_bounds__(BLOCK)
void hmm_forward_wmma(const int* __restrict__ ys,
                      const float* __restrict__ transition,
                      const float* __restrict__ emission,
                      const float* __restrict__ choice,
                      const float* __restrict__ prior,
                      float* __restrict__ out)
{
    __shared__ float     alpha[NCOMP];              // log alpha, composite states
    __shared__ float     gnewP[KCH*GP_STRIDE];      // padded 3 x 16 x 144 WMMA output
    __shared__ _Float16  betaQ[KCH*BQ_STRIDE];      // 3 fragment-major copies, K padded
    __shared__ float     elog[KCH*NS*ALPH];         // log_softmax(emission)
    __shared__ _Float16  texpH[KCH*NS*NS];          // exp(log_softmax(transition))[q][s_old][s_new]
    __shared__ float     plog[KCH*NS];              // log_softmax(prior)
    __shared__ float     cst[KCH];                  // log_softmax(choice)
    __shared__ int       ysl[TLEN];
    __shared__ float     wred[32];                  // per-wave reduction partials

    const int tid  = threadIdx.x;
    const int lane = tid & 31;
    const int wv   = tid >> 5;      // 0..26
    const int q    = wv / 9;        // chain that transitions
    const int ct   = wv % 9;        // 16-wide column tile (144 = 9*16)

    // ---------------- precompute (log-softmax tables) ----------------
    if (tid < TLEN) ysl[tid] = ys[tid];
    for (int i = tid; i < KCH*BQ_STRIDE; i += BLOCK) betaQ[i] = (_Float16)0.0f; // K>=12 pad stays 0
    if (tid == 0) {
        float mx = fmaxf(choice[0], fmaxf(choice[1], choice[2]));
        float s  = __expf(choice[0]-mx) + __expf(choice[1]-mx) + __expf(choice[2]-mx);
        float l  = mx + __logf(s);
        cst[0] = choice[0]-l; cst[1] = choice[1]-l; cst[2] = choice[2]-l;
    }
    if (tid < KCH*NS) {             // 36 transition rows of 12
        int qq = tid / NS, so = tid % NS;
        const float* row = transition + (qq*NS + so)*NS;
        float mx = -INFINITY;
        for (int j = 0; j < NS; ++j) mx = fmaxf(mx, row[j]);
        float s = 0.f;
        for (int j = 0; j < NS; ++j) s += __expf(row[j]-mx);
        float inv = 1.0f / s;
        for (int j = 0; j < NS; ++j)
            texpH[(qq*NS+so)*NS + j] = (_Float16)(__expf(row[j]-mx)*inv);
    }
    if (tid < KCH*NS) {             // 36 emission rows of 32
        int qq = tid / NS, s0 = tid % NS;
        const float* row = emission + (qq*NS + s0)*ALPH;
        float mx = -INFINITY;
        for (int j = 0; j < ALPH; ++j) mx = fmaxf(mx, row[j]);
        float s = 0.f;
        for (int j = 0; j < ALPH; ++j) s += __expf(row[j]-mx);
        float l = mx + __logf(s);
        for (int j = 0; j < ALPH; ++j) elog[(qq*NS+s0)*ALPH + j] = row[j]-l;
    }
    if (tid < KCH) {                // 3 prior rows of 12
        const float* row = prior + tid*NS;
        float mx = -INFINITY;
        for (int j = 0; j < NS; ++j) mx = fmaxf(mx, row[j]);
        float s = 0.f;
        for (int j = 0; j < NS; ++j) s += __expf(row[j]-mx);
        float l = mx + __logf(s);
        for (int j = 0; j < NS; ++j) plog[tid*NS+j] = row[j]-l;
    }
    __syncthreads();

    // ---- loop-invariant per-lane A fragment (exp(t_q)^T, 16x32, zero-padded) ----
    const int nrow = lane & 15;              // A row (s_new) / B col lane / C col lane
    const int kb   = (lane >> 4) << 3;       // K offset for upper half-wave
    const int col  = ct*16 + nrow;           // 0..143
    v16h afrag;
    #pragma unroll
    for (int e = 0; e < 16; ++e) {
        int k = ((e >> 3) << 4) + kb + (e & 7);   // 0..31
        afrag[e] = (k < NS && nrow < NS) ? texpH[q*144 + k*12 + nrow] : (_Float16)0.0f;
    }
    const _Float16* bq = &betaQ[q*BQ_STRIDE + col*32 + kb];  // 16B aligned
    float* gp = &gnewP[q*GP_STRIDE + col];

    // ---- loop-invariant per-thread descriptors ----
    // alpha-update: this thread owns composite states i = tid + r*BLOCK, r=0..5
    int   upd_g[6]; int upd_e[6]; float upd_c[6]; float p0[6];
    #pragma unroll
    for (int r = 0; r < 6; ++r) {
        int i  = tid + r*BLOCK;
        int ch = i % 3; int rr = i / 3;
        int s2 = rr % 12, s1 = (rr/12) % 12, s0 = rr/144;
        int sn, colIdx;
        if      (ch == 0) { sn = s0; colIdx = s1*12 + s2; }
        else if (ch == 1) { sn = s1; colIdx = s0*12 + s2; }
        else              { sn = s2; colIdx = s0*12 + s1; }
        upd_g[r] = ch*GP_STRIDE + sn*144 + colIdx;
        upd_e[r] = (ch*NS + sn)*ALPH;
        upd_c[r] = cst[ch];
        p0[r]    = plog[s0] + plog[12+s1] + plog[24+s2] + cst[ch];
    }
    // beta scatter: this thread owns tuples u = tid + r*BLOCK, r=0..1
    int bo0[2], bo1[2], bo2[2];
    #pragma unroll
    for (int r = 0; r < 2; ++r) {
        int u  = tid + r*BLOCK;
        int s2 = u % 12, s1 = (u/12) % 12, s0 = u/144;
        bo0[r] = (s1*12 + s2)*32 + s0;
        bo1[r] = BQ_STRIDE   + (s0*12 + s2)*32 + s1;
        bo2[r] = 2*BQ_STRIDE + (s0*12 + s1)*32 + s2;
    }

    // alpha0
    #pragma unroll
    for (int r = 0; r < 6; ++r) alpha[tid + r*BLOCK] = p0[r];
    __syncthreads();

    // ---------------- 64 sequential forward steps ----------------
    for (int t = 0; t < TLEN; ++t) {
        // running max m = max(alpha): wave shfl-reduce + one LDS stage
        float lm = -INFINITY;
        #pragma unroll
        for (int r = 0; r < 6; ++r) lm = fmaxf(lm, alpha[tid + r*BLOCK]);
        lm = waveMax(lm);
        if (lane == 0) wred[wv] = lm;
        __syncthreads();                                   // (A)
        float m = waveMax(wred[lane < 27 ? lane : 0]);

        // betaP scatter into 3 fragment-major padded layouts
        #pragma unroll
        for (int r = 0; r < 2; ++r) {
            int u = tid + r*BLOCK;
            float s = __expf(alpha[3*u+0]-m) + __expf(alpha[3*u+1]-m) + __expf(alpha[3*u+2]-m);
            _Float16 h = (_Float16)s;
            betaQ[bo0[r]] = h; betaQ[bo1[r]] = h; betaQ[bo2[r]] = h;
        }
        __syncthreads();                                   // (B)

        // one v_wmma tile per wave: B = two unconditional b128 LDS loads
        v8h blo = *(const v8h*)(bq);
        v8h bhi = *(const v8h*)(bq + 16);
        v16h b  = __builtin_shufflevector(blo, bhi,
                    0,1,2,3,4,5,6,7,8,9,10,11,12,13,14,15);
        v8f cacc = {};
        cacc = __builtin_amdgcn_wmma_f32_16x16x32_f16(
                   false, afrag, false, b, (short)0, cacc, false, false);
        const int mroff = kb;   // C rows v+8 for upper half-wave
        #pragma unroll
        for (int v = 0; v < 8; ++v) gp[(v + mroff)*144] = cacc[v];
        __syncthreads();                                   // (C)

        // alpha_new = Blog + c + m + log(G)
        int y = ysl[t];
        #pragma unroll
        for (int r = 0; r < 6; ++r) {
            float g = gnewP[upd_g[r]];
            alpha[tid + r*BLOCK] = elog[upd_e[r] + y] + (upd_c[r] + m) + __logf(g);
        }
        __syncthreads();                                   // (D)
    }

    // ---------------- final logsumexp(alpha) ----------------
    float lm = -INFINITY;
    #pragma unroll
    for (int r = 0; r < 6; ++r) lm = fmaxf(lm, alpha[tid + r*BLOCK]);
    lm = waveMax(lm);
    if (lane == 0) wred[wv] = lm;
    __syncthreads();
    float m2 = waveMax(wred[lane < 27 ? lane : 0]);
    __syncthreads();

    float ls = 0.f;
    #pragma unroll
    for (int r = 0; r < 6; ++r) ls += __expf(alpha[tid + r*BLOCK] - m2);
    ls = waveSum(ls);
    if (lane == 0) wred[wv] = ls;
    __syncthreads();
    if (tid == 0) {
        float s = 0.f;
        for (int w = 0; w < 27; ++w) s += wred[w];
        out[0] = m2 + __logf(s);
    }
}

extern "C" void kernel_launch(void* const* d_in, const int* in_sizes, int n_in,
                              void* d_out, int out_size, void* d_ws, size_t ws_size,
                              hipStream_t stream) {
    (void)in_sizes; (void)n_in; (void)out_size; (void)d_ws; (void)ws_size;
    const int*   ys         = (const int*)  d_in[0];
    const float* transition = (const float*)d_in[1];
    const float* emission   = (const float*)d_in[2];
    const float* choice     = (const float*)d_in[3];
    const float* prior      = (const float*)d_in[4];
    float*       out        = (float*)d_out;
    hipLaunchKernelGGL(hmm_forward_wmma, dim3(1), dim3(BLOCK), 0, stream,
                       ys, transition, emission, choice, prior, out);
}
